// CrossAttention_2611340116354
// MI455X (gfx1250) — compile-verified, matrix operand order checked by reading, not run on hardware
//
#include <hip/hip_runtime.h>
#include <hip/hip_bf16.h>
#include <math.h>

typedef __attribute__((ext_vector_type(16))) _Float16 v16h;
typedef __attribute__((ext_vector_type(8)))  float    v8f;

#define NUM_HEADS 8
#define DIM_EMBED 640
#define DIM_CROSS 768
#define DIM_HEAD  80
#define SEQ_Q     1024
#define SEQ_KV    77
#define BATCH     32

// Load one 16x32 f16 fragment from LDS (row-major, element stride).
// ISA layout (CDNA5 ISA, 16-bit A 16x32): lanes 0-15 -> M=0..15,
// VGPR0..3 = K 0..7, VGPR4..7 = K 16..23; lanes 16-31 same M, K shifted by 8.
__device__ __forceinline__ v16h frag_ld(const _Float16* __restrict__ base,
                                        int row, int col, int stride, int lane) {
  const _Float16* p = base + (size_t)(row + (lane & 15)) * stride + col + ((lane & 16) ? 8 : 0);
  v16h f;
#pragma unroll
  for (int i = 0; i < 8; ++i) { f[i] = p[i]; f[i + 8] = p[i + 16]; }
  return f;
}

// C/D 16x16 f32 fragment store: lane -> N = lane%16, VGPR j -> M = j (+8 lanes>=16)
template <typename OT>
__device__ __forceinline__ void frag_st(OT* __restrict__ C, const v8f& c,
                                        const float* __restrict__ bias,
                                        int M, int N, int mBase, int nBase, int lane) {
  const int col = nBase + (lane & 15);
  const int rb  = (lane & 16) ? 8 : 0;
  const float bv = bias[col];
#pragma unroll
  for (int j = 0; j < 8; ++j) {
    const int row = mBase + rb + j;
    if (row < M) C[(size_t)row * N + col] = (OT)(c[j] + bv);
  }
}

// ---------------------------------------------------------------------------
// All-f16 GEMM with transposed B:  C[M,N] = A[M,K] @ Bt[N,K]^T + bias[N]
// BM=128, BN=128, BK=32; 8 waves, each owns a 32x64 tile (8 WMMAs/iter).
// Double-buffered LDS with register staging: global loads of tile k+1 are in
// flight while WMMAs run on tile k; one __syncthreads per K-step.
// ---------------------------------------------------------------------------
template <typename OT>
__global__ __launch_bounds__(256)
void wmma_gemm_f16(const _Float16* __restrict__ A,
                   const _Float16* __restrict__ Bt,
                   const float* __restrict__ bias,
                   OT* __restrict__ C,
                   int M, int N, int K) {
  constexpr int BM = 128, BN = 128, BK = 32, LDT = BK + 8;  // 40 f16 = 20 banks
  __shared__ _Float16 As[2][BM * LDT];
  __shared__ _Float16 Bs[2][BN * LDT];

  const int tid  = threadIdx.x;
  const int lane = tid & 31;
  const int wid  = tid >> 5;
  const int m0 = blockIdx.y * BM;
  const int n0 = blockIdx.x * BN;
  const int wm = (wid & 3) * 32;   // wave row offset (4 waves along M)
  const int wn = (wid >> 2) * 64;  // wave col offset (2 waves along N)

  // Staging map: each thread owns 16 contiguous f16 of one row (2 x uint4).
  const int srow  = tid >> 1;          // 0..127
  const int khalf = (tid & 1) * 16;    // 0 or 16
  const int gm = m0 + srow;
  const int gn = n0 + srow;
  const uint4 zero4 = make_uint4(0u, 0u, 0u, 0u);

  uint4 aR0, aR1, bR0, bR1;
  // prologue: stage tile k0 = 0
  {
    if (gm < M) {
      const uint4* ap = (const uint4*)(A + (size_t)gm * K + khalf);
      aR0 = ap[0]; aR1 = ap[1];
    } else { aR0 = zero4; aR1 = zero4; }
    if (gn < N) {
      const uint4* bp = (const uint4*)(Bt + (size_t)gn * K + khalf);
      bR0 = bp[0]; bR1 = bp[1];
    } else { bR0 = zero4; bR1 = zero4; }
    uint4* asd = (uint4*)&As[0][srow * LDT + khalf];
    asd[0] = aR0; asd[1] = aR1;
    uint4* bsd = (uint4*)&Bs[0][srow * LDT + khalf];
    bsd[0] = bR0; bsd[1] = bR1;
  }

  v8f acc[2][4];
#pragma unroll
  for (int i = 0; i < 2; ++i)
#pragma unroll
    for (int j = 0; j < 4; ++j) acc[i][j] = (v8f){};

  int cur = 0;
  for (int k0 = 0; k0 < K; k0 += BK) {
    __syncthreads();
    const int kn = k0 + BK;
    const bool more = kn < K;
    if (more) {  // issue next tile's global loads (latency hidden by WMMAs)
      if (gm < M) {
        const uint4* ap = (const uint4*)(A + (size_t)gm * K + kn + khalf);
        aR0 = ap[0]; aR1 = ap[1];
      } else { aR0 = zero4; aR1 = zero4; }
      if (gn < N) {
        const uint4* bp = (const uint4*)(Bt + (size_t)gn * K + kn + khalf);
        bR0 = bp[0]; bR1 = bp[1];
      } else { bR0 = zero4; bR1 = zero4; }
    }

    const _Float16* Ac = As[cur];
    const _Float16* Bc = Bs[cur];
    const v16h a0 = frag_ld(Ac, wm,      0, LDT, lane);
    const v16h a1 = frag_ld(Ac, wm + 16, 0, LDT, lane);
    const v16h b0 = frag_ld(Bc, wn,      0, LDT, lane);
    const v16h b1 = frag_ld(Bc, wn + 16, 0, LDT, lane);
    const v16h b2 = frag_ld(Bc, wn + 32, 0, LDT, lane);
    const v16h b3 = frag_ld(Bc, wn + 48, 0, LDT, lane);
    acc[0][0] = __builtin_amdgcn_wmma_f32_16x16x32_f16(false, a0, false, b0, (short)0, acc[0][0], false, false);
    acc[0][1] = __builtin_amdgcn_wmma_f32_16x16x32_f16(false, a0, false, b1, (short)0, acc[0][1], false, false);
    acc[0][2] = __builtin_amdgcn_wmma_f32_16x16x32_f16(false, a0, false, b2, (short)0, acc[0][2], false, false);
    acc[0][3] = __builtin_amdgcn_wmma_f32_16x16x32_f16(false, a0, false, b3, (short)0, acc[0][3], false, false);
    acc[1][0] = __builtin_amdgcn_wmma_f32_16x16x32_f16(false, a1, false, b0, (short)0, acc[1][0], false, false);
    acc[1][1] = __builtin_amdgcn_wmma_f32_16x16x32_f16(false, a1, false, b1, (short)0, acc[1][1], false, false);
    acc[1][2] = __builtin_amdgcn_wmma_f32_16x16x32_f16(false, a1, false, b2, (short)0, acc[1][2], false, false);
    acc[1][3] = __builtin_amdgcn_wmma_f32_16x16x32_f16(false, a1, false, b3, (short)0, acc[1][3], false, false);

    if (more) {  // stage into the other buffer (no sync needed: disjoint buffer)
      const int nxt = cur ^ 1;
      uint4* asd = (uint4*)&As[nxt][srow * LDT + khalf];
      asd[0] = aR0; asd[1] = aR1;
      uint4* bsd = (uint4*)&Bs[nxt][srow * LDT + khalf];
      bsd[0] = bR0; bsd[1] = bR1;
      cur = nxt;
    }
  }

#pragma unroll
  for (int i = 0; i < 2; ++i)
#pragma unroll
    for (int j = 0; j < 4; ++j)
      frag_st(C, acc[i][j], bias, M, N, m0 + wm + i * 16, n0 + wn + j * 16, lane);
}

// ---------------------------------------------------------------------------
// Fused attention per (batch, head, 32 q-rows): scores -> softmax -> P@V.
// ---------------------------------------------------------------------------
__global__ __launch_bounds__(256)
void attention_kernel(const _Float16* __restrict__ q,
                      const _Float16* __restrict__ k,
                      const _Float16* __restrict__ v,
                      _Float16* __restrict__ o) {
  constexpr int QT = 32;        // q rows per block
  constexpr int KP = 96;        // padded inner dims (d=80 -> 96, kv=77 -> 96)
  constexpr int LD = KP + 8;    // 104 f16 = 52 banks -> conflict-free
  constexpr int NP = 80;        // padded kv for score columns (5 n-tiles)
  __shared__ _Float16 Qs[QT * LD];        // reused as softmax-weight matrix
  __shared__ _Float16 Ks[NP * LD];        // Ks[kv][d]
  __shared__ _Float16 Vt[DIM_HEAD * LD];  // Vt[d][kv]
  __shared__ float    Sc[QT * NP];

  const int tid  = threadIdx.x;
  const int lane = tid & 31;
  const int wid  = tid >> 5;
  constexpr int QTILES = SEQ_Q / QT;
  const int bh = blockIdx.x / QTILES;
  const int qt = blockIdx.x % QTILES;
  const int b  = bh / NUM_HEADS;
  const int h  = bh % NUM_HEADS;
  const size_t qbase  = ((size_t)b * SEQ_Q + (size_t)qt * QT) * DIM_EMBED + h * DIM_HEAD;
  const size_t kvbase = (size_t)b * SEQ_KV * DIM_EMBED + h * DIM_HEAD;

  for (int idx = tid; idx < QT * KP; idx += 256) {
    const int r = idx / KP, c = idx % KP;
    Qs[r * LD + c] = (c < DIM_HEAD) ? q[qbase + (size_t)r * DIM_EMBED + c] : (_Float16)0.0f;
  }
  for (int idx = tid; idx < NP * KP; idx += 256) {
    const int kv = idx / KP, d = idx % KP;
    Ks[kv * LD + d] = (kv < SEQ_KV && d < DIM_HEAD)
        ? k[kvbase + (size_t)kv * DIM_EMBED + d] : (_Float16)0.0f;
  }
  for (int idx = tid; idx < DIM_HEAD * KP; idx += 256) {
    const int d = idx / KP, kv = idx % KP;
    Vt[d * LD + kv] = (kv < SEQ_KV)
        ? v[kvbase + (size_t)kv * DIM_EMBED + d] : (_Float16)0.0f;
  }
  __syncthreads();

  // GEMM1: Sc[QT][NP] = (Qs @ Ks^T) * 1/sqrt(D)
  const float scale = 0.11180339887498949f;  // 1/sqrt(80)
  constexpr int NT1 = NP / 16;               // 5
  for (int t = wid; t < (QT / 16) * NT1; t += 8) {
    const int mt = (t / NT1) * 16;
    const int nt = (t % NT1) * 16;
    v8f s = {};
#pragma unroll
    for (int ks = 0; ks < KP; ks += 32) {
      const v16h a  = frag_ld(Qs, mt, ks, LD, lane);
      const v16h bf = frag_ld(Ks, nt, ks, LD, lane);
      s = __builtin_amdgcn_wmma_f32_16x16x32_f16(false, a, false, bf, (short)0, s, false, false);
    }
    const int col = nt + (lane & 15);
    const int rb  = (lane & 16) ? 8 : 0;
#pragma unroll
    for (int j = 0; j < 8; ++j)
      Sc[(mt + rb + j) * NP + col] = s[j] * scale;
  }
  __syncthreads();

  // Row softmax over 77 valid columns; write f16 weights into Qs (reuse), pad->0.
  if (tid < QT) {
    float* row = &Sc[tid * NP];
    float mx = -1e30f;
    for (int c = 0; c < SEQ_KV; ++c) mx = fmaxf(mx, row[c]);
    float s = 0.0f;
    for (int c = 0; c < SEQ_KV; ++c) { const float e = __expf(row[c] - mx); row[c] = e; s += e; }
    const float inv = 1.0f / s;
    _Float16* wr = &Qs[tid * LD];
    for (int c = 0; c < SEQ_KV; ++c) wr[c] = (_Float16)(row[c] * inv);
    for (int c = SEQ_KV; c < KP; ++c) wr[c] = (_Float16)0.0f;
  }
  __syncthreads();

  // GEMM2: O[QT][80] = W[QT][96] @ V[96][80]  (Vt holds B transposed)
  constexpr int NT2 = DIM_HEAD / 16;  // 5
  for (int t = wid; t < (QT / 16) * NT2; t += 8) {
    const int mt = (t / NT2) * 16;
    const int nt = (t % NT2) * 16;
    v8f s = {};
#pragma unroll
    for (int ks = 0; ks < KP; ks += 32) {
      const v16h a  = frag_ld(Qs, mt, ks, LD, lane);
      const v16h bf = frag_ld(Vt, nt, ks, LD, lane);
      s = __builtin_amdgcn_wmma_f32_16x16x32_f16(false, a, false, bf, (short)0, s, false, false);
    }
    const int col = nt + (lane & 15);
    const int rb  = (lane & 16) ? 8 : 0;
#pragma unroll
    for (int j = 0; j < 8; ++j)
      o[qbase + (size_t)(mt + rb + j) * DIM_EMBED + col] = (_Float16)s[j];
  }
}

// ---------------------------------------------------------------------------
// Pre-pass conversions
// ---------------------------------------------------------------------------
__global__ __launch_bounds__(256)
void cvt_f16(const float* __restrict__ in, _Float16* __restrict__ out, int n) {
  const int i = (blockIdx.x * 256 + threadIdx.x) * 4;
  if (i + 3 < n) {
    const float4 v = *(const float4*)(in + i);
    out[i + 0] = (_Float16)v.x;
    out[i + 1] = (_Float16)v.y;
    out[i + 2] = (_Float16)v.z;
    out[i + 3] = (_Float16)v.w;
  } else {
    for (int j = i; j < n; ++j) out[j] = (_Float16)in[j];
  }
}

// Wt[n][k] = (f16) W[k][n]  for W stored [K][N] row-major.
__global__ __launch_bounds__(256)
void cvt_transpose(const float* __restrict__ W, _Float16* __restrict__ Wt,
                   int K, int N) {
  const int idx = blockIdx.x * 256 + threadIdx.x;
  if (idx >= K * N) return;
  const int kk = idx / N, nn = idx % N;
  Wt[(size_t)nn * K + kk] = (_Float16)W[idx];
}

extern "C" void kernel_launch(void* const* d_in, const int* in_sizes, int n_in,
                              void* d_out, int out_size, void* d_ws, size_t ws_size,
                              hipStream_t stream) {
  (void)in_sizes; (void)n_in; (void)out_size; (void)ws_size;
  const float* x  = (const float*)d_in[0];
  const float* y  = (const float*)d_in[1];
  const float* Wq = (const float*)d_in[2];
  const float* bq = (const float*)d_in[3];
  const float* Wk = (const float*)d_in[4];
  const float* bk = (const float*)d_in[5];
  const float* Wv = (const float*)d_in[6];
  const float* bv = (const float*)d_in[7];
  const float* Wo = (const float*)d_in[8];
  const float* bo = (const float*)d_in[9];
  float* out = (float*)d_out;

  const int Mq  = BATCH * SEQ_Q;   // 32768
  const int Mkv = BATCH * SEQ_KV;  // 2464
  const int E   = DIM_EMBED;
  const int CK  = DIM_CROSS;

  // Workspace layout (f16). aws aliases x16: x16 is dead after the q-projection.
  char* ws = (char*)d_ws;
  size_t off = 0;
  _Float16* x16  = (_Float16*)(ws + off); off += (size_t)Mq  * E  * 2;  // also aws
  _Float16* q16  = (_Float16*)(ws + off); off += (size_t)Mq  * E  * 2;
  _Float16* y16  = (_Float16*)(ws + off); off += (size_t)Mkv * CK * 2;
  _Float16* k16  = (_Float16*)(ws + off); off += (size_t)Mkv * E  * 2;
  _Float16* v16  = (_Float16*)(ws + off); off += (size_t)Mkv * E  * 2;
  _Float16* WqT  = (_Float16*)(ws + off); off += (size_t)E  * E * 2;
  _Float16* WkT  = (_Float16*)(ws + off); off += (size_t)CK * E * 2;
  _Float16* WvT  = (_Float16*)(ws + off); off += (size_t)CK * E * 2;
  _Float16* WoT  = (_Float16*)(ws + off); off += (size_t)E  * E * 2;
  _Float16* aws  = x16;

  const dim3 blk(256);

  // Pre-pass: convert activations, convert+transpose weights (all tiny vs GEMMs).
  cvt_f16<<<dim3((Mq * E / 4 + 255) / 256), blk, 0, stream>>>(x, x16, Mq * E);
  cvt_f16<<<dim3((Mkv * CK / 4 + 255) / 256), blk, 0, stream>>>(y, y16, Mkv * CK);
  cvt_transpose<<<dim3((E * E + 255) / 256), blk, 0, stream>>>(Wq, WqT, E, E);
  cvt_transpose<<<dim3((CK * E + 255) / 256), blk, 0, stream>>>(Wk, WkT, CK, E);
  cvt_transpose<<<dim3((CK * E + 255) / 256), blk, 0, stream>>>(Wv, WvT, CK, E);
  cvt_transpose<<<dim3((E * E + 255) / 256), blk, 0, stream>>>(Wo, WoT, E, E);

  // Projections (all-f16 WMMA GEMMs, f32 accumulate).
  wmma_gemm_f16<_Float16>
      <<<dim3(E / 128, (Mq + 127) / 128), blk, 0, stream>>>(x16, WqT, bq, q16, Mq, E, E);
  wmma_gemm_f16<_Float16>
      <<<dim3(E / 128, (Mkv + 127) / 128), blk, 0, stream>>>(y16, WkT, bk, k16, Mkv, E, CK);
  wmma_gemm_f16<_Float16>
      <<<dim3(E / 128, (Mkv + 127) / 128), blk, 0, stream>>>(y16, WvT, bv, v16, Mkv, E, CK);

  // Fused attention (scores -> softmax -> P@V), all WMMA. Writes aws (= old x16).
  attention_kernel<<<dim3(BATCH * NUM_HEADS * (SEQ_Q / 32)), blk, 0, stream>>>(q16, k16, v16, aws);

  // Output projection: out = attn @ Wo + bo (f16 in, f32 out).
  wmma_gemm_f16<float>
      <<<dim3(E / 128, (Mq + 127) / 128), blk, 0, stream>>>(aws, WoT, bo, out, Mq, E, E);
}
